// AutoEncoder_84516366450763
// MI455X (gfx1250) — matrix operand order repaired and checked by reading, not compiled, not run on hardware
//
#include <hip/hip_runtime.h>
#include <stdint.h>

#define BATCH 4096
#define EMBED 768
#define FEATS 24576
#define TOPK  64
#define FSPLIT 4                  // feature chunks (grid.y)
#define FCHUNK (FEATS / FSPLIT)   // 6144 columns per chunk
#define MROWS 64                  // batch rows per workgroup
#define CAP   128                 // 64 kept + <=64 appends per 64-col round
#define AROW  (EMBED + 8)         // padded LDS row stride (bf16 elems)
#define KSTEPS (EMBED / 32)       // 24 WMMA K-steps

typedef __attribute__((ext_vector_type(16))) __bf16 v16bf;
typedef __attribute__((ext_vector_type(8)))  float  v8f;

union FragBF { v16bf v; uint4 q[2]; };

__device__ __forceinline__ unsigned short f2bf(float x) {
  union { float f; uint32_t u; } c; c.f = x;
  uint32_t r = c.u + 0x7FFFu + ((c.u >> 16) & 1u);   // RNE
  return (unsigned short)(r >> 16);
}
__device__ __forceinline__ uint32_t pack2bf(float lo, float hi) {
  return (uint32_t)f2bf(lo) | ((uint32_t)f2bf(hi) << 16);
}

// ---------------- K0: enc_weight f32 -> bf16, packed fragment-major ----------------
// Block b = ((tile*24 + kk)*32 + lane): 32 B = 16 bf16 = W[tile*16 + (lane&15)]
// [kk*32 + (lane>>4)*16 .. +15].  A wave's B read per kk step is one contiguous 1 KB run.
__global__ __launch_bounds__(256) void pack_w_bf16(const float* __restrict__ w,
                                                   unsigned short* __restrict__ wb) {
  size_t b = (size_t)blockIdx.x * 256 + threadIdx.x;   // 1,179,648 blocks total
  int lane = (int)(b & 31);
  size_t tk = b >> 5;                                  // tile*24 + kk
  int kk = (int)(tk % KSTEPS);
  size_t tile = tk / KSTEPS;
  int row = (int)tile * 16 + (lane & 15);
  int k0  = kk * 32 + (lane >> 4) * 16;
  const float* src = w + (size_t)row * EMBED + k0;
  float4 f0 = *(const float4*)(src + 0);
  float4 f1 = *(const float4*)(src + 4);
  float4 f2 = *(const float4*)(src + 8);
  float4 f3 = *(const float4*)(src + 12);
  uint4 o0, o1;
  o0.x = pack2bf(f0.x, f0.y); o0.y = pack2bf(f0.z, f0.w);
  o0.z = pack2bf(f1.x, f1.y); o0.w = pack2bf(f1.z, f1.w);
  o1.x = pack2bf(f2.x, f2.y); o1.y = pack2bf(f2.z, f2.w);
  o1.z = pack2bf(f3.x, f3.y); o1.w = pack2bf(f3.z, f3.w);
  uint4* dst = (uint4*)(wb + b * 16);
  dst[0] = o0;
  dst[1] = o1;
}

// Wave-cooperative: move top-64 of buf[0..cnt) to front (descending). 32 lanes.
template <typename IDX>
__device__ __forceinline__ void select_topk(volatile float* v, volatile IDX* id,
                                            int cnt, int lane) {
  int keep = cnt < TOPK ? cnt : TOPK;
  for (int i = 0; i < keep; ++i) {
    float best = -3.4e38f; int bestj = -1;
    for (int j = i + lane; j < cnt; j += 32) {
      float x = v[j];
      if (x > best) { best = x; bestj = j; }
    }
    for (int off = 16; off; off >>= 1) {
      float ob = __shfl_xor(best, off, 32);
      int   oj = __shfl_xor(bestj, off, 32);
      if (ob > best) { best = ob; bestj = oj; }
    }
    if (lane == 0 && bestj > i) {
      float tv = v[i];  v[i]  = v[bestj];  v[bestj]  = tv;
      IDX   ti = id[i]; id[i] = id[bestj]; id[bestj] = ti;
    }
    __builtin_amdgcn_wave_barrier();
  }
}

// ---------------- K1: encoder WMMA GEMM + per-chunk top-64 ----------------
// grid (64, 4): blockIdx.x = 64-row block, blockIdx.y = feature chunk.
// 512 threads = 16 waves: wave = (m_tile 0..3) x (n_tile 0..3); 64 cols/round.
__global__ __launch_bounds__(512) void sae_topk(const float* __restrict__ embed,
                                                const float* __restrict__ enc_bias,
                                                const unsigned short* __restrict__ wb,
                                                float* __restrict__ cv,
                                                int* __restrict__ ci) {
  __shared__ unsigned short Atile[MROWS * AROW];        // 99.3 KB
  __shared__ float          svals[MROWS][CAP];          // 32.8 KB
  __shared__ unsigned short sidxs[MROWS][CAP];          // 16.4 KB
  __shared__ __align__(16) float sthresh[MROWS];
  __shared__ int            scnt[MROWS];

  const int tid  = threadIdx.x;
  const int lane = tid & 31;
  const int wid  = tid >> 5;                 // 16 waves
  const int row0 = blockIdx.x * MROWS;
  const int fc0  = blockIdx.y * FCHUNK;

  if (tid < MROWS) { scnt[tid] = 0; sthresh[tid] = -3.4e38f; }
  for (int i = tid; i < MROWS * EMBED; i += 512) {
    int r = i / EMBED;
    int c = i - r * EMBED;
    float x = embed[(size_t)(row0 + r) * EMBED + c] - enc_bias[c];
    Atile[r * AROW + c] = f2bf(x);
  }
  __syncthreads();

  const int halfsel = lane >> 4;             // 0: lanes 0-15, 1: lanes 16-31
  const int nsel    = lane & 15;
  const int m_tile  = wid >> 2;              // 0..3 -> rows m_tile*16..+15
  const int n_tile  = wid & 3;               // 0..3 -> 16-col slice of round
  const int rbase   = m_tile * 16 + halfsel * 8;
  // A fragment (16x32 bf16): elems 0-15 = K k0+16*halfsel..k0+16*halfsel+15
  const unsigned short* arow = Atile + (m_tile * 16 + nsel) * AROW + halfsel * 16;

  for (int fbase = 0; fbase < FCHUNK; fbase += 64) {    // 96 rounds
    const int f0 = fc0 + fbase + n_tile * 16;
    // packed-B base for this 16-col tile: per-lane 32 B blocks, kk stride = 512 elems
    const unsigned short* wtile =
        wb + ((size_t)(f0 >> 4) * (KSTEPS * 32) + lane) * 16;
    // prefetch this wave's B tile for the next round (+4 tiles = 96 KB)
    __builtin_prefetch(wtile + 4 * (KSTEPS * 32) * 16, 0, 1);

    // thresholds stable within a round: cache 8 row-thresholds in regs
    float th[8];
    *(float4*)&th[0] = *(const float4*)&sthresh[rbase];
    *(float4*)&th[4] = *(const float4*)&sthresh[rbase + 4];

    v8f acc = {0.f, 0.f, 0.f, 0.f, 0.f, 0.f, 0.f, 0.f};
#pragma unroll 4
    for (int kk = 0; kk < KSTEPS; ++kk) {               // 24 WMMA steps, K=768
      FragBF a, b;
      a.q[0] = *(const uint4*)(arow + kk * 32);
      a.q[1] = *(const uint4*)(arow + kk * 32 + 8);
      b.q[0] = *(const uint4*)(wtile + kk * 512);
      b.q[1] = *(const uint4*)(wtile + kk * 512 + 8);
      acc = __builtin_amdgcn_wmma_f32_16x16x32_bf16(false, a.v, false, b.v,
                                                    (short)0, acc, false, false);
    }
    // C layout: vgpr j -> row = rbase + j, col = f0 + nsel
#pragma unroll
    for (int j = 0; j < 8; ++j) {
      float vsc = acc[j];
      if (vsc > th[j]) {
        int r = rbase + j;
        int pos = atomicAdd(&scnt[r], 1);
        if (pos < CAP) { svals[r][pos] = vsc; sidxs[r][pos] = (unsigned short)(f0 + nsel); }
      }
    }
    __syncthreads();
    // compaction: wave w owns rows 4w..4w+3; trigger keeps cnt <= CAP invariant
    for (int rr = wid * 4; rr < wid * 4 + 4; ++rr) {
      int c = scnt[rr]; if (c > CAP) c = CAP;
      if (c > TOPK) {
        select_topk(&svals[rr][0], &sidxs[rr][0], c, lane);
        if (lane == 0) { scnt[rr] = TOPK; sthresh[rr] = svals[rr][TOPK - 1]; }
      }
    }
    __syncthreads();
  }

  // finalize: exactly top-64 per row for this chunk -> workspace
  for (int rr = wid * 4; rr < wid * 4 + 4; ++rr) {
    int c = scnt[rr]; if (c > CAP) c = CAP;
    if (c > TOPK) select_topk(&svals[rr][0], &sidxs[rr][0], c, lane);
    size_t base = (size_t)(row0 + rr) * (FSPLIT * TOPK) + (size_t)blockIdx.y * TOPK;
    for (int j = lane; j < TOPK; j += 32) {
      cv[base + j] = svals[rr][j];
      ci[base + j] = (int)sidxs[rr][j];
    }
  }
}

// ---------------- K2: merge 4x64 candidates -> top-64, decoder, normalize ----------------
__global__ __launch_bounds__(256) void merge_decode(const float* __restrict__ cv,
                                                    const int* __restrict__ ci,
                                                    const float* __restrict__ enc_bias,
                                                    const float* __restrict__ dec,
                                                    float* __restrict__ out) {
  __shared__ float mv[16][FSPLIT * TOPK];
  __shared__ int   mi[16][FSPLIT * TOPK];
  __shared__ float sred[8];
  __shared__ float sscale;

  const int tid  = threadIdx.x;
  const int lane = tid & 31;
  const int wid  = tid >> 5;
  const int row0 = blockIdx.x * 16;

  for (int i = tid; i < 16 * FSPLIT * TOPK; i += 256) {
    int r = i >> 8;           // FSPLIT*TOPK == 256
    int j = i & 255;
    mv[r][j] = cv[(size_t)(row0 + r) * 256 + j];
    mi[r][j] = ci[(size_t)(row0 + r) * 256 + j];
  }
  __syncthreads();

  for (int rr = wid * 2; rr < wid * 2 + 2; ++rr)
    select_topk(&mv[rr][0], &mi[rr][0], FSPLIT * TOPK, lane);
  __syncthreads();

  // decoder gather + bias, L2 normalize; 768 = 3 * 256 dims per thread
  const int e0 = tid, e1 = tid + 256, e2 = tid + 512;
  for (int r = 0; r < 16; ++r) {
    float a0 = enc_bias[e0], a1 = enc_bias[e1], a2 = enc_bias[e2];
    for (int j = 0; j < TOPK; ++j) {
      float w = mv[r][j];
      const float* dv = dec + (size_t)mi[r][j] * EMBED;
      a0 = fmaf(w, dv[e0], a0);
      a1 = fmaf(w, dv[e1], a1);
      a2 = fmaf(w, dv[e2], a2);
    }
    float ss = a0 * a0 + a1 * a1 + a2 * a2;
    for (int off = 16; off; off >>= 1) ss += __shfl_xor(ss, off, 32);
    if (lane == 0) sred[wid] = ss;
    __syncthreads();
    if (tid == 0) {
      float t = 0.f;
#pragma unroll
      for (int i = 0; i < 8; ++i) t += sred[i];
      float n = sqrtf(t);
      sscale = 1.0f / (n > 1e-12f ? n : 1e-12f);
    }
    __syncthreads();
    float s = sscale;
    float* o = out + (size_t)(row0 + r) * EMBED;
    o[e0] = a0 * s; o[e1] = a1 * s; o[e2] = a2 * s;
    __syncthreads();
  }
}

extern "C" void kernel_launch(void* const* d_in, const int* in_sizes, int n_in,
                              void* d_out, int out_size, void* d_ws, size_t ws_size,
                              hipStream_t stream) {
  (void)in_sizes; (void)n_in; (void)out_size; (void)ws_size;
  const float* embed    = (const float*)d_in[0];
  const float* enc_bias = (const float*)d_in[1];
  const float* enc_w    = (const float*)d_in[2];
  const float* dec      = (const float*)d_in[3];
  float* outp           = (float*)d_out;

  // workspace layout
  unsigned short* wbf = (unsigned short*)d_ws;                       // 37,748,736 B (packed)
  float* cv = (float*)((char*)d_ws + (size_t)FEATS * EMBED * 2);     // 4,194,304 B
  int*   ci = (int*)((char*)cv + (size_t)BATCH * FSPLIT * TOPK * 4); // 4,194,304 B

  const int npack = (FEATS / 16) * KSTEPS * 32 / 256;                // 4608 blocks
  pack_w_bf16<<<npack, 256, 0, stream>>>(enc_w, wbf);
  sae_topk<<<dim3(BATCH / MROWS, FSPLIT), 512, 0, stream>>>(embed, enc_bias, wbf, cv, ci);
  merge_decode<<<BATCH / 16, 256, 0, stream>>>(cv, ci, enc_bias, dec, outp);
}
